// Eff_TTEmbedding_29901562314965
// MI455X (gfx1250) — matrix operand order repaired and checked by reading, not compiled, not run on hardware
//
#include <hip/hip_runtime.h>
#include <stdint.h>

// ---------------------------------------------------------------------------
// TT-embedding, MI455X (gfx1250, wave32).
//   out[s] = core0[i0] (4x32)  @  W[i1,i2] (32x32),   W = contract(core1,core2)
//   (i1,i2) pair index pi = idx % 10000  (only 10,000 distinct W matrices)
// Pipeline: histogram -> scan(+pad bins to 4) -> scatter (counting sort by pi,
//           entries packed (i0<<20)|sample_id) -> build W f16 (transposed
//           [n][k]) -> grouped GEMM: 2x V_WMMA_F32_16X16X32_F16 per 4 samples,
//           100% tile utilization, NT stores for the 512 MB output stream.
// Workspace use: ~26 MB.
// ---------------------------------------------------------------------------

typedef __attribute__((ext_vector_type(16))) _Float16 v16h;
typedef __attribute__((ext_vector_type(8)))  _Float16 v8h;
typedef __attribute__((ext_vector_type(8)))  float    v8f;

#define NBINS 10000
#define SENT  0xFFFFFFFFu
#define IDMASK 0xFFFFFu

// workspace byte offsets
static constexpr size_t WS_COUNTS  = 0;               // 10000 u32
static constexpr size_t WS_OFFSETS = 64u  * 1024;     // 10000 u32
static constexpr size_t WS_CURSORS = 128u * 1024;     // 10000 u32
static constexpr size_t WS_META    = 192u * 1024;     // meta[0] = nChunks
static constexpr size_t WS_SORTED  = 256u * 1024;     // up to n + 4*NBINS u32
static constexpr size_t WS_A16     = 5u * 1024 * 1024;  // 100*128 f16
static constexpr size_t WS_W16     = 6u * 1024 * 1024;  // 10000*1024 f16 (20 MB)

__global__ void zero_counts(uint32_t* __restrict__ counts, uint32_t* __restrict__ meta) {
  int i = blockIdx.x * blockDim.x + threadIdx.x;
  if (i < NBINS) counts[i] = 0u;
  if (i < 16) meta[i] = 0u;
}

__global__ void hist_kernel(const int* __restrict__ idx, int n,
                            uint32_t* __restrict__ counts) {
  int i = blockIdx.x * blockDim.x + threadIdx.x;
  if (i < n) atomicAdd(&counts[(uint32_t)idx[i] % 10000u], 1u);
}

// single workgroup: pad each bin to a multiple of 4, exclusive scan, emit
// offsets + cursors + chunk count.
__global__ void scan_kernel(const uint32_t* __restrict__ counts,
                            uint32_t* __restrict__ offsets,
                            uint32_t* __restrict__ cursors,
                            uint32_t* __restrict__ meta) {
  __shared__ uint32_t part[1024];
  const int tid = threadIdx.x;
  const int base = tid * 10;
  uint32_t local[10];
  uint32_t sum = 0;
  #pragma unroll
  for (int j = 0; j < 10; ++j) {
    int b = base + j;
    uint32_t c = (b < NBINS) ? ((counts[b] + 3u) & ~3u) : 0u;
    local[j] = sum;           // exclusive within thread
    sum += c;
  }
  part[tid] = sum;
  __syncthreads();
  for (int off = 1; off < 1024; off <<= 1) {
    uint32_t v = 0;
    if (tid >= off) v = part[tid - off];
    __syncthreads();
    part[tid] += v;
    __syncthreads();
  }
  uint32_t tbase = part[tid] - sum;   // exclusive prefix of thread sums
  #pragma unroll
  for (int j = 0; j < 10; ++j) {
    int b = base + j;
    if (b < NBINS) {
      uint32_t o = tbase + local[j];
      offsets[b] = o;
      cursors[b] = o;
    }
  }
  if (tid == 1023) meta[0] = part[1023] >> 2;   // number of 4-sample chunks
}

__global__ void fill_sentinel(uint32_t* __restrict__ sorted, int cap) {
  for (int i = blockIdx.x * blockDim.x + threadIdx.x; i < cap;
       i += gridDim.x * blockDim.x)
    sorted[i] = SENT;
}

// entries packed (i0 << 20) | sample_id  (id < 2^20, i0 < 100)
__global__ void scatter_kernel(const int* __restrict__ idx, int n,
                               uint32_t* __restrict__ cursors,
                               uint32_t* __restrict__ sorted) {
  int i = blockIdx.x * blockDim.x + threadIdx.x;
  if (i < n) {
    uint32_t v  = (uint32_t)idx[i];
    uint32_t pi = v % 10000u;
    uint32_t i0 = v / 10000u;
    uint32_t pos = atomicAdd(&cursors[pi], 1u);
    sorted[pos] = (i0 << 20) | (uint32_t)i;
  }
}

__global__ void conv_a_kernel(const float* __restrict__ core0,
                              _Float16* __restrict__ A16) {
  int i = blockIdx.x * blockDim.x + threadIdx.x;
  if (i < 100 * 128) A16[i] = (_Float16)core0[i];
}

// W16[p][n][k] (transposed for contiguous B-fragment loads):
//   W[k=r1][n=q1*4+q2] = sum_r2 core1[i1][r1*256 + q1*32 + r2] * core2[i2][r2*4 + q2]
__global__ void build_w_kernel(const float* __restrict__ core1,
                               const float* __restrict__ core2,
                               _Float16* __restrict__ W16) {
  const int p  = blockIdx.x;           // 0..9999
  const int i1 = p / 100, i2 = p % 100;
  const float* b = core1 + (size_t)i1 * 8192;  // [r1][q1][r2]
  __shared__ float cs[128];                    // core2 row [r2][q2]
  if (threadIdx.x < 128) cs[threadIdx.x] = core2[(size_t)i2 * 128 + threadIdx.x];
  __syncthreads();
  for (int o = threadIdx.x; o < 1024; o += blockDim.x) {
    int n = o >> 5, k = o & 31;
    int q1 = n >> 2, q2 = n & 3;
    const float* brow = b + k * 256 + q1 * 32;   // 32 consecutive r2 values
    float s = 0.f;
    #pragma unroll
    for (int r2 = 0; r2 < 32; ++r2) s += brow[r2] * cs[r2 * 4 + q2];
    W16[(size_t)p * 1024 + o] = (_Float16)s;
  }
}

// Grouped GEMM: each 4-sample chunk -> A[16x32] (4 samples x 4 q0 rows, shared
// B = W tile of the chunk's bin) -> two V_WMMA_F32_16X16X32_F16 (N tiles).
__global__ void __launch_bounds__(256)
tt_gemm_kernel(const int* __restrict__ indices,
               const _Float16* __restrict__ A16,
               const _Float16* __restrict__ W16,
               const uint32_t* __restrict__ sorted,
               const uint32_t* __restrict__ meta,
               float* __restrict__ out) {
  const int lane = threadIdx.x & 31;
  const int wpb  = blockDim.x >> 5;
  const uint32_t w   = blockIdx.x * wpb + (threadIdx.x >> 5);
  const uint32_t nw  = gridDim.x * wpb;
  const uint32_t nChunks = meta[0];
  uint64_t beg = ((uint64_t)nChunks * w) / nw;
  uint64_t end = ((uint64_t)nChunks * (w + 1)) / nw;

  // ISA 7.12.2 wave32 layouts
  const int mrow  = lane & 15;         // A row index (M)
  const int jA    = mrow >> 2;         // which sample of the chunk this row is
  const int q0A   = mrow & 3;
  const int khalf = (lane >> 4) & 1;   // A and B: K 0-15 / 16-31
  const int ncol  = lane & 15;         // B/C column (N)
  const bool hiHalf = lane >= 16;

  uint32_t lastPi = SENT;
  v16h bf0{}, bf1{};

  for (uint64_t c = beg; c < end; ++c) {
    const uint32_t* sl = sorted + (c << 2);
    uint4 idv = *(const uint4*)sl;     // 4 packed (i0<<20)|id, front-packed
    const uint32_t ref = idv.x;        // always valid (pad < 4 per bin)
    const uint32_t pi  = (uint32_t)indices[ref & IDMASK] % 10000u; // uniform

    if (pi != lastPi) {   // B fragments: column ncol, 16 contiguous K halfs
      const _Float16* wt = W16 + (size_t)pi * 1024 + khalf * 16;
      bf0 = *(const v16h*)(wt + (size_t)ncol * 32);
      bf1 = *(const v16h*)(wt + (size_t)(ncol + 16) * 32);
      lastPi = pi;
    }

    // A fragment: row mrow = (sample jA, q0A); i0 comes packed in the entry
    uint32_t pj = (jA & 2) ? ((jA & 1) ? idv.w : idv.z)
                           : ((jA & 1) ? idv.y : idv.x);
    uint32_t psafe = (pj == SENT) ? ref : pj;
    uint32_t i0 = psafe >> 20;
    const _Float16* ar = A16 + (size_t)i0 * 128 + q0A * 32 + khalf * 16;
    v8h alo = *(const v8h*)(ar);        // K kh*16 .. +7
    v8h ahi = *(const v8h*)(ar + 8);    // K kh*16+8 .. +15
    v16h af;
    #pragma unroll
    for (int t = 0; t < 8; ++t) { af[t] = alo[t]; af[t + 8] = ahi[t]; }

    v8f cz = {};
    v8f c0 = __builtin_amdgcn_wmma_f32_16x16x32_f16(false, af, false, bf0,
                                                    (short)0, cz, false, false);
    v8f c1 = __builtin_amdgcn_wmma_f32_16x16x32_f16(false, af, false, bf1,
                                                    (short)0, cz, false, false);

    // C: lane holds column ncol; low half rows 0-7 (samples 0,1),
    // high half rows 8-15 (samples 2,3). Two samples per lane -> two branches.
    uint32_t pA = hiHalf ? idv.z : idv.x;   // rows 0-3 of this half (c[0..3])
    uint32_t pB = hiHalf ? idv.w : idv.y;   // rows 4-7 of this half (c[4..7])
    if (pA != SENT) {
      float* base = out + (size_t)(pA & IDMASK) * 128 + ncol;
      #pragma unroll
      for (int t = 0; t < 4; ++t) {
        __builtin_nontemporal_store(c0[t], base + t * 32);
        __builtin_nontemporal_store(c1[t], base + t * 32 + 16);
      }
    }
    if (pB != SENT) {
      float* base = out + (size_t)(pB & IDMASK) * 128 + ncol;
      #pragma unroll
      for (int t = 0; t < 4; ++t) {
        __builtin_nontemporal_store(c0[t + 4], base + t * 32);
        __builtin_nontemporal_store(c1[t + 4], base + t * 32 + 16);
      }
    }
  }
}

extern "C" void kernel_launch(void* const* d_in, const int* in_sizes, int n_in,
                              void* d_out, int out_size, void* d_ws, size_t ws_size,
                              hipStream_t stream) {
  const int*   indices = (const int*)d_in[0];
  const float* core0   = (const float*)d_in[1];
  const float* core1   = (const float*)d_in[2];
  const float* core2   = (const float*)d_in[3];
  const int n = in_sizes[0];

  uint8_t* ws = (uint8_t*)d_ws;
  uint32_t* counts  = (uint32_t*)(ws + WS_COUNTS);
  uint32_t* offsets = (uint32_t*)(ws + WS_OFFSETS);
  uint32_t* cursors = (uint32_t*)(ws + WS_CURSORS);
  uint32_t* meta    = (uint32_t*)(ws + WS_META);
  uint32_t* sorted  = (uint32_t*)(ws + WS_SORTED);
  _Float16* A16     = (_Float16*)(ws + WS_A16);
  _Float16* W16     = (_Float16*)(ws + WS_W16);
  const int cap = n + 4 * NBINS;   // max padded total

  zero_counts   <<<(NBINS + 255) / 256, 256, 0, stream>>>(counts, meta);
  hist_kernel   <<<(n + 255) / 256,     256, 0, stream>>>(indices, n, counts);
  scan_kernel   <<<1,                  1024, 0, stream>>>(counts, offsets, cursors, meta);
  fill_sentinel <<<1024,                256, 0, stream>>>(sorted, cap);
  scatter_kernel<<<(n + 255) / 256,     256, 0, stream>>>(indices, n, cursors, sorted);
  conv_a_kernel <<<(100 * 128 + 255) / 256, 256, 0, stream>>>(core0, A16);
  build_w_kernel<<<NBINS,               256, 0, stream>>>(core1, core2, W16);
  tt_gemm_kernel<<<1024,                256, 0, stream>>>(indices, A16, W16,
                                                          sorted, meta, (float*)d_out);
}
